// UpsampleBlock_31473520345762
// MI455X (gfx1250) — matrix operand order repaired and checked by reading, not compiled, not run on hardware
//
#include <hip/hip_runtime.h>
#include <hip/hip_bf16.h>

// ---------------- constants ----------------
#define BB   4
#define NN   4096
#define SS   1024
#define C1   128
#define C2   256
#define CIN  (C1 + C2)   // 384
#define CO   128
#define KNN  16
#define NPTS (BB * NN)   // 16384

typedef __attribute__((ext_vector_type(16))) __bf16 v16bf;
typedef __attribute__((ext_vector_type(8)))  float  v8f;

union FragBF {
    v16bf        v;
    unsigned int u[8];
    uint4        q[2];
};

// ---------------- bf16 helpers (storage = ushort bits) ----------------
__device__ __forceinline__ unsigned short f2bf(float f) {
    unsigned u = __builtin_bit_cast(unsigned, f);
    u += 0x7FFFu + ((u >> 16) & 1u);          // round-to-nearest-even
    return (unsigned short)(u >> 16);
}
__device__ __forceinline__ unsigned pack2(float lo, float hi) {
    return (unsigned)f2bf(lo) | ((unsigned)f2bf(hi) << 16);
}
__device__ __forceinline__ float bfbits2f(unsigned hbits) {
    unsigned u = hbits << 16;
    return __builtin_bit_cast(float, u);
}

// ============================================================
// K0: one-time fp32 -> bf16 weight packing (fuse_w 128x384, lu_w 128x128)
// ============================================================
__global__ __launch_bounds__(256)
void pack_weights_kernel(const float* __restrict__ fw, const float* __restrict__ lw,
                         unsigned short* __restrict__ fwb, unsigned short* __restrict__ lwb)
{
    const int id = blockIdx.x * 256 + threadIdx.x;
    if (id < CO * CIN) fwb[id] = f2bf(fw[id]);
    if (id < CO * CO)  lwb[id] = f2bf(lw[id]);
}

// ============================================================
// K1: 3-point interpolation (top-3 LARGEST dists, faithful) +
//     concat -> bf16 new_pts [NPTS, 384]
// ============================================================
__global__ __launch_bounds__(256)
void interp_kernel(const float* __restrict__ xyz1, const float* __restrict__ xyz2,
                   const float* __restrict__ points1, const float* __restrict__ points2,
                   unsigned short* __restrict__ newpts)
{
    __shared__ float s2[SS * 3];                       // 12 KB
    const int tid = threadIdx.x;
    const int gid = blockIdx.x * 256 + tid;            // point id over B*N
    const int b   = gid >> 12;
    const int n   = gid & (NN - 1);

    for (int i = tid; i < SS * 3; i += 256)
        s2[i] = xyz2[(size_t)b * SS * 3 + i];
    __syncthreads();

    const float px = xyz1[(size_t)gid * 3 + 0];
    const float py = xyz1[(size_t)gid * 3 + 1];
    const float pz = xyz1[(size_t)gid * 3 + 2];

    float d0 = -1e30f, d1 = -1e30f, d2 = -1e30f;
    int   i0 = 0,      i1 = 0,      i2 = 0;
    for (int s = 0; s < SS; ++s) {
        const float dx = px - s2[s * 3 + 0];
        const float dy = py - s2[s * 3 + 1];
        const float dz = pz - s2[s * 3 + 2];
        const float d  = dx * dx + dy * dy + dz * dz;
        if (d > d2) {
            if (d > d0)      { d2 = d1; i2 = i1; d1 = d0; i1 = i0; d0 = d; i0 = s; }
            else if (d > d1) { d2 = d1; i2 = i1; d1 = d;  i1 = s; }
            else             { d2 = d;  i2 = s; }
        }
    }
    const float r0 = 1.0f / (d0 + 1e-8f);
    const float r1 = 1.0f / (d1 + 1e-8f);
    const float r2 = 1.0f / (d2 + 1e-8f);
    const float rs = r0 + r1 + r2;
    const float w0 = r0 / rs, w1 = r1 / rs, w2 = r2 / rs;

    const size_t row = (size_t)gid * CIN;
    // skip-connection channels 0..127 (points1 is [B,C1,N]: coalesced in n)
    for (int c = 0; c < C1; ++c)
        newpts[row + c] = f2bf(points1[((size_t)b * C1 + c) * NN + n]);
    // interpolated channels 128..383
    const float* __restrict__ P2 = points2 + (size_t)b * C2 * SS;
    for (int c = 0; c < C2; ++c) {
        const float v = w0 * P2[(size_t)c * SS + i0]
                      + w1 * P2[(size_t)c * SS + i1]
                      + w2 * P2[(size_t)c * SS + i2];
        newpts[row + C1 + c] = f2bf(v);
    }
}

// ============================================================
// K2: KNN(k=16) over self distances, per-batch indices
// ============================================================
__global__ __launch_bounds__(256)
void knn_kernel(const float* __restrict__ xyz1, int* __restrict__ knn)
{
    __shared__ float sx[1024 * 3];                     // 12 KB chunk
    const int tid = threadIdx.x;
    const int gid = blockIdx.x * 256 + tid;
    const int b   = gid >> 12;

    const float px = xyz1[(size_t)gid * 3 + 0];
    const float py = xyz1[(size_t)gid * 3 + 1];
    const float pz = xyz1[(size_t)gid * 3 + 2];

    float nnd[KNN]; int nni[KNN];
#pragma unroll
    for (int j = 0; j < KNN; ++j) { nnd[j] = 1e30f; nni[j] = 0; }
    float worst = 1e30f; int wslot = 0;

    for (int chunk = 0; chunk < 4; ++chunk) {
        __syncthreads();
        for (int i = tid; i < 1024 * 3; i += 256)
            sx[i] = xyz1[((size_t)b * NN + chunk * 1024) * 3 + i];
        __syncthreads();
        for (int s = 0; s < 1024; ++s) {
            const float dx = px - sx[s * 3 + 0];
            const float dy = py - sx[s * 3 + 1];
            const float dz = pz - sx[s * 3 + 2];
            const float d  = dx * dx + dy * dy + dz * dz;
            if (d < worst) {
                nnd[wslot] = d; nni[wslot] = chunk * 1024 + s;
                worst = nnd[0]; wslot = 0;
#pragma unroll
                for (int j = 1; j < KNN; ++j)
                    if (nnd[j] > worst) { worst = nnd[j]; wslot = j; }
            }
        }
    }
    for (int j = 0; j < KNN; ++j)
        knn[(size_t)gid * KNN + j] = nni[j];
}

// ============================================================
// K3/K7: bf16 WMMA GEMM  Out[b,co,n] = W[co,:K] . Brows[n,:K] (+bias, opt relu)
//   Wb  bf16 [128,KD]   (pre-packed)
//   Brows bf16 [NPTS,KD]
//   Out fp32 [B,CO,N]
//   one wave = one 16(co) tile, iterates NT 16-point n-tiles reusing the
//   A fragments held in registers. Per k-step: one batched clause of
//   2*NT global_load_b128, then NT back-to-back v_wmma_f32_16x16x32_bf16
//   with staggered load waits; next-step lines prefetched into WGP$.
// ============================================================
template <int KD, int NT, int RELU>
__global__ __launch_bounds__(128)
void wmma_gemm_kernel(const unsigned short* __restrict__ Wb,
                      const unsigned short* __restrict__ Brows,
                      const float* __restrict__ bias, float* __restrict__ Out)
{
    constexpr int KSTEPS = KD / 32;
    const int lane   = threadIdx.x & 31;
    const int wave   = threadIdx.x >> 5;
    const int mtile  = blockIdx.y * 4 + wave;      // 0..7
    const int cobase = mtile * 16;
    const int m      = lane & 15;                  // row of A / col of B
    const int khalf  = lane >> 4;

    // ---- load all A fragments once (reused for NT n-tiles) ----
    // A (16x32 bf16) layout: lane holds row M=m; VGPR v covers
    //   K = (v>=4?16:0) + khalf*8 + (v&3)*2  -> two contiguous 16B runs.
    FragBF a[KSTEPS];
    const unsigned short* __restrict__ wrow = Wb + (size_t)(cobase + m) * KD;
#pragma unroll
    for (int ks = 0; ks < KSTEPS; ++ks) {
        a[ks].q[0] = *(const uint4*)(wrow + ks * 32 + khalf * 8);
        a[ks].q[1] = *(const uint4*)(wrow + ks * 32 + 16 + khalf * 8);
    }

    // B (32x16 bf16) layout: lane holds column N=m; VGPR v covers
    //   K = (v>=4?16:0) + khalf*8 + (v&3)*2  -> two contiguous 16B runs.
    const unsigned short* brow[NT];
    v8f acc[NT] = {};
#pragma unroll
    for (int t = 0; t < NT; ++t)
        brow[t] = Brows + (size_t)((blockIdx.x * NT + t) * 16 + m) * KD + khalf * 8;

#pragma unroll
    for (int ks = 0; ks < KSTEPS; ++ks) {
        FragBF bm[NT];
#pragma unroll
        for (int t = 0; t < NT; ++t) {               // one 2*NT-load clause
            bm[t].q[0] = *(const uint4*)(brow[t] + ks * 32);
            bm[t].q[1] = *(const uint4*)(brow[t] + ks * 32 + 16);
        }
        if (ks + 1 < KSTEPS) {
#pragma unroll
            for (int t = 0; t < NT; ++t)             // pull next step into WGP$
                __builtin_prefetch(brow[t] + (ks + 1) * 32, 0, 0);
        }
#pragma unroll
        for (int t = 0; t < NT; ++t)                 // NT back-to-back WMMAs
            acc[t] = __builtin_amdgcn_wmma_f32_16x16x32_bf16(
                         false, a[ks].v, false, bm[t].v, (short)0, acc[t], false, false);
    }

    // C/D layout: VGPR r -> M = r + 8*khalf ; lane%16 = N
#pragma unroll
    for (int t = 0; t < NT; ++t) {
        const int nglob = (blockIdx.x * NT + t) * 16 + m;
        const int b_    = nglob >> 12;
        const int n_    = nglob & (NN - 1);
#pragma unroll
        for (int r = 0; r < 8; ++r) {
            const int co = cobase + r + khalf * 8;
            float v = acc[t][r] + bias[co];
            if (RELU) v = fmaxf(v, 0.0f);
            Out[((size_t)b_ * CO + co) * NN + n_] = v;
        }
    }
}

// ============================================================
// K4/K8: per-channel batch stats -> folded scale/shift
// ============================================================
__global__ __launch_bounds__(256)
void bn_stats_kernel(const float* __restrict__ X, const float* __restrict__ gamma,
                     const float* __restrict__ beta, float* __restrict__ scale,
                     float* __restrict__ shift)
{
    __shared__ float ssum[256], ssq[256];
    const int co = blockIdx.x, tid = threadIdx.x;
    float s = 0.0f, q = 0.0f;
    for (int i = tid; i < NPTS; i += 256) {
        const int b = i >> 12, n = i & (NN - 1);
        const float v = X[((size_t)b * CO + co) * NN + n];
        s += v; q += v * v;
    }
    ssum[tid] = s; ssq[tid] = q; __syncthreads();
    for (int st = 128; st > 0; st >>= 1) {
        if (tid < st) { ssum[tid] += ssum[tid + st]; ssq[tid] += ssq[tid + st]; }
        __syncthreads();
    }
    if (tid == 0) {
        const float inv  = 1.0f / (float)NPTS;
        const float mean = ssum[0] * inv;
        const float var  = ssq[0] * inv - mean * mean;
        const float sc   = gamma[co] * rsqrtf(var + 1e-5f);
        scale[co] = sc;
        shift[co] = beta[co] - mean * sc;
    }
}

// ============================================================
// K5: BN+ReLU apply + bf16 point-major transpose
// ============================================================
__global__ __launch_bounds__(256)
void bn_apply_fuse_kernel(const float* __restrict__ X, const float* __restrict__ scale,
                          const float* __restrict__ shift, float* __restrict__ Xbn,
                          unsigned short* __restrict__ Xt)
{
    const int id = blockIdx.x * 256 + threadIdx.x;   // 0..B*CO*N-1
    const int n  = id & (NN - 1);
    const int co = (id >> 12) & (CO - 1);
    const int b  = id >> 19;
    float v = X[id] * scale[co] + shift[co];
    v = fmaxf(v, 0.0f);
    Xbn[id] = v;
    Xt[((size_t)(b * NN + n)) * CO + co] = f2bf(v);
}

// ============================================================
// K6: Laplacian gather: dx[n,c] = sum_knn xt[idx,c] - xt[n,c] (bf16 out)
//     one wave per point; lane covers 4 channels (coalesced 256B rows)
// ============================================================
__global__ __launch_bounds__(128)
void lap_gather_kernel(const unsigned short* __restrict__ Xt, const int* __restrict__ knn,
                       unsigned short* __restrict__ Dx)
{
    const int lane = threadIdx.x & 31;
    const int wave = threadIdx.x >> 5;
    const int row  = blockIdx.x * 4 + wave;          // point id
    const int b    = row >> 12;
    const int c0   = lane * 4;

    const unsigned short* __restrict__ selfrow = Xt + (size_t)row * CO;
    const unsigned su0 = *(const unsigned*)(selfrow + c0);
    const unsigned su1 = *(const unsigned*)(selfrow + c0 + 2);

    float a0 = 0.0f, a1 = 0.0f, a2 = 0.0f, a3 = 0.0f;
    for (int j = 0; j < KNN; ++j) {
        const int idx = knn[(size_t)row * KNN + j];
        const unsigned short* __restrict__ nr = Xt + ((size_t)(b * NN) + idx) * CO;
        const unsigned u0 = *(const unsigned*)(nr + c0);
        const unsigned u1 = *(const unsigned*)(nr + c0 + 2);
        a0 += bfbits2f(u0 & 0xFFFFu); a1 += bfbits2f(u0 >> 16);
        a2 += bfbits2f(u1 & 0xFFFFu); a3 += bfbits2f(u1 >> 16);
    }
    a0 -= bfbits2f(su0 & 0xFFFFu); a1 -= bfbits2f(su0 >> 16);
    a2 -= bfbits2f(su1 & 0xFFFFu); a3 -= bfbits2f(su1 >> 16);

    unsigned* __restrict__ drow = (unsigned*)(Dx + (size_t)row * CO + c0);
    drow[0] = pack2(a0, a1);
    drow[1] = pack2(a2, a3);
}

// ============================================================
// K9: residual: out = x_bn + (relu_h * scale2 + shift2)
// ============================================================
__global__ __launch_bounds__(256)
void final_kernel(const float* __restrict__ Xbn, const float* __restrict__ R,
                  const float* __restrict__ scale2, const float* __restrict__ shift2,
                  float* __restrict__ out)
{
    const int id = blockIdx.x * 256 + threadIdx.x;
    const int co = (id >> 12) & (CO - 1);
    out[id] = Xbn[id] + R[id] * scale2[co] + shift2[co];
}

// ============================================================
// launcher
// ============================================================
extern "C" void kernel_launch(void* const* d_in, const int* in_sizes, int n_in,
                              void* d_out, int out_size, void* d_ws, size_t ws_size,
                              hipStream_t stream)
{
    const float* xyz1       = (const float*)d_in[0];
    const float* xyz2       = (const float*)d_in[1];
    const float* points1    = (const float*)d_in[2];
    const float* points2    = (const float*)d_in[3];
    const float* fuse_w     = (const float*)d_in[4];
    const float* fuse_b     = (const float*)d_in[5];
    const float* fuse_gamma = (const float*)d_in[6];
    const float* fuse_beta  = (const float*)d_in[7];
    const float* lu_w       = (const float*)d_in[8];
    const float* lu_b       = (const float*)d_in[9];
    const float* lu_gamma   = (const float*)d_in[10];
    const float* lu_beta    = (const float*)d_in[11];

    char* ws = (char*)d_ws;
    unsigned short* newpts = (unsigned short*)ws; ws += (size_t)NPTS * CIN * 2;  // 12.6 MB
    float*          X      = (float*)ws;          ws += (size_t)NPTS * CO * 4;   //  8.4 MB
    float*          Xbn    = (float*)ws;          ws += (size_t)NPTS * CO * 4;   //  8.4 MB
    unsigned short* Xt     = (unsigned short*)ws; ws += (size_t)NPTS * CO * 2;   //  4.2 MB
    int*            knn    = (int*)ws;            ws += (size_t)NPTS * KNN * 4;  //  1.0 MB
    unsigned short* Dx     = (unsigned short*)ws; ws += (size_t)NPTS * CO * 2;   //  4.2 MB
    float*          R      = (float*)ws;          ws += (size_t)NPTS * CO * 4;   //  8.4 MB
    unsigned short* fwb    = (unsigned short*)ws; ws += (size_t)CO * CIN * 2;    //  96 KB
    unsigned short* lwb    = (unsigned short*)ws; ws += (size_t)CO * CO * 2;     //  32 KB
    float*          scale1 = (float*)ws;          ws += CO * 4;
    float*          shift1 = (float*)ws;          ws += CO * 4;
    float*          scale2 = (float*)ws;          ws += CO * 4;
    float*          shift2 = (float*)ws;          ws += CO * 4;

    // 0. one-time weight packing to bf16 (removes all cvt VALU from GEMM loop)
    pack_weights_kernel<<<(CO * CIN) / 256, 256, 0, stream>>>(fuse_w, lu_w, fwb, lwb);
    // 1. interpolation + concat -> bf16 activations
    interp_kernel<<<NPTS / 256, 256, 0, stream>>>(xyz1, xyz2, points1, points2, newpts);
    // 2. KNN indices
    knn_kernel<<<NPTS / 256, 256, 0, stream>>>(xyz1, knn);
    // 3. fuse 1x1 conv via WMMA (K=384), no relu (BN comes first)
    wmma_gemm_kernel<CIN, 4, 0><<<dim3(NPTS / 16 / 4, 2), 128, 0, stream>>>(fwb, newpts, fuse_b, X);
    // 4. BN1 stats
    bn_stats_kernel<<<CO, 256, 0, stream>>>(X, fuse_gamma, fuse_beta, scale1, shift1);
    // 5. BN1 + ReLU apply, emit fp32 [B,CO,N] + bf16 [NPTS,CO]
    bn_apply_fuse_kernel<<<(NPTS * CO) / 256, 256, 0, stream>>>(X, scale1, shift1, Xbn, Xt);
    // 6. Laplacian neighbor-sum minus self -> bf16 dx rows
    lap_gather_kernel<<<NPTS / 4, 128, 0, stream>>>(Xt, knn, Dx);
    // 7. lu 1x1 conv via WMMA (K=128) + bias + relu (relu BEFORE bn, faithful)
    wmma_gemm_kernel<CO, 4, 1><<<dim3(NPTS / 16 / 4, 2), 128, 0, stream>>>(lwb, Dx, lu_b, R);
    // 8. BN2 stats over relu'd activations
    bn_stats_kernel<<<CO, 256, 0, stream>>>(R, lu_gamma, lu_beta, scale2, shift2);
    // 9. residual add -> d_out [B,128,4096] fp32
    final_kernel<<<(NPTS * CO) / 256, 256, 0, stream>>>(Xbn, R, scale2, shift2, (float*)d_out);
}
